// DeepseekMoEGate_21388937134645
// MI455X (gfx1250) — compile-verified, matrix operand order checked by reading, not run on hardware
//
#include <hip/hip_runtime.h>
#include <hip/hip_bf16.h>

typedef __attribute__((ext_vector_type(16))) __bf16 v16bf;
typedef __attribute__((ext_vector_type(8)))  float  v8f;

#define T_TOK   16384
#define NEXP    64
#define HDIM    2048
#define TOPK    6
#define SEQLEN  4096
#define NBATCH  4
#define ALPHA   0.001f

#define NKCHUNK (HDIM / 32)            // 64 K-chunks of 32
#define BFRAG   (32 * 16)              // bf16 elems per (chunk, n-tile) B fragment

// Load 16 f32 per lane in WMMA 16-bit A/B fragment order (elements 0..7 = p[0..7],
// elements 8..15 = p[16..23]) and split each f32 into bf16 hi + bf16 lo residual.
__device__ __forceinline__ void load_cvt_tile(const float* __restrict__ p,
                                              v16bf& hi, v16bf& lo) {
  const float4* p4 = (const float4*)p;
  float4 a = p4[0], b = p4[1], c = p4[4], d = p4[5];
  float f[16] = {a.x, a.y, a.z, a.w, b.x, b.y, b.z, b.w,
                 c.x, c.y, c.z, c.w, d.x, d.y, d.z, d.w};
#pragma unroll
  for (int i = 0; i < 16; ++i) {
    __bf16 h = (__bf16)f[i];
    hi[i] = h;
    lo[i] = (__bf16)(f[i] - (float)h);
  }
}

// One-shot: split W[64,2048] into bf16 hi/lo, stored in WMMA B-fragment order:
// wp[((kc*4 + nt)*32 + lane)*16 + i]  -> per-lane 32 B contiguous, coalesced.
__global__ __launch_bounds__(256) void gate_pack_w(const float* __restrict__ w,
                                                   __bf16* __restrict__ wp_hi,
                                                   __bf16* __restrict__ wp_lo) {
  const int tid  = blockIdx.x * 256 + threadIdx.x;   // 8192 = 64 chunks * 4 tiles * 32 lanes
  const int lane = tid & 31;
  const int nt   = (tid >> 5) & 3;
  const int kc   = tid >> 7;
  const int r    = lane & 15;
  const int off  = (lane >> 4) * 8;
  const float* wr = w + (size_t)(nt * 16 + r) * HDIM + kc * 32 + off;
  v16bf hi, lo;
  load_cvt_tile(wr, hi, lo);
  *(v16bf*)(wp_hi + (size_t)tid * 16) = hi;
  *(v16bf*)(wp_lo + (size_t)tid * 16) = lo;
}

// logits[T,64] = x @ W^T via bf16 hi/lo split WMMA (fp32-accurate accumulate).
// One wave handles TWO 16-token M-tiles (32 tokens) to halve B read traffic;
// B comes pre-split/pre-swizzled from the pack kernel (no conversion in loop).
__global__ __launch_bounds__(128) void gate_gemm_wmma(const float* __restrict__ x,
                                                      const __bf16* __restrict__ wp_hi,
                                                      const __bf16* __restrict__ wp_lo,
                                                      float* __restrict__ logits) {
  const int lane = threadIdx.x & 31;
  const int wave = threadIdx.x >> 5;
  const int tile = blockIdx.x * 4 + wave;   // 32-token tile
  const int r    = lane & 15;
  const int off  = (lane >> 4) * 8;         // lanes 16-31 take K offsets +8 / +24
  const int row0 = tile * 32;

  const float* xp0 = x + (size_t)(row0 + r) * HDIM + off;
  const float* xp1 = xp0 + (size_t)16 * HDIM;

  v8f c0[4] = {{}, {}, {}, {}};
  v8f c1[4] = {{}, {}, {}, {}};

  for (int kb = 0; kb < HDIM; kb += 32) {
    v16bf ah0, al0, ah1, al1;
    load_cvt_tile(xp0 + kb, ah0, al0);
    load_cvt_tile(xp1 + kb, ah1, al1);

    const size_t bbase = ((size_t)(kb >> 5) * 4) * BFRAG + (size_t)lane * 16;
#pragma unroll
    for (int nt = 0; nt < 4; ++nt) {
      v16bf bh = *(const v16bf*)(wp_hi + bbase + (size_t)nt * BFRAG);
      v16bf bl = *(const v16bf*)(wp_lo + bbase + (size_t)nt * BFRAG);
      c0[nt] = __builtin_amdgcn_wmma_f32_16x16x32_bf16(false, ah0, false, bh, (short)0, c0[nt], false, false);
      c0[nt] = __builtin_amdgcn_wmma_f32_16x16x32_bf16(false, ah0, false, bl, (short)0, c0[nt], false, false);
      c0[nt] = __builtin_amdgcn_wmma_f32_16x16x32_bf16(false, al0, false, bh, (short)0, c0[nt], false, false);
      c1[nt] = __builtin_amdgcn_wmma_f32_16x16x32_bf16(false, ah1, false, bh, (short)0, c1[nt], false, false);
      c1[nt] = __builtin_amdgcn_wmma_f32_16x16x32_bf16(false, ah1, false, bl, (short)0, c1[nt], false, false);
      c1[nt] = __builtin_amdgcn_wmma_f32_16x16x32_bf16(false, al1, false, bh, (short)0, c1[nt], false, false);
    }
  }

  // C/D layout: lanes 0-15 -> M = vr, lanes 16-31 -> M = vr + 8; N = lane & 15.
  float* outp = logits + (size_t)row0 * NEXP;
#pragma unroll
  for (int vr = 0; vr < 8; ++vr) {
    const int m = vr + (off ? 8 : 0);
#pragma unroll
    for (int nt = 0; nt < 4; ++nt) {
      outp[(size_t)m * NEXP + nt * 16 + r]        = c0[nt][vr];
      outp[(size_t)(m + 16) * NEXP + nt * 16 + r] = c1[nt][vr];
    }
  }
}

__global__ void gate_init(float* p) { p[threadIdx.x + blockIdx.x * blockDim.x] = 0.f; }

// One wave32 per token: softmax over 64 experts (2/lane), iterative top-6 argmax
// with lower-index tie-break, renormalize; accumulate per-(batch,expert) stats.
__global__ __launch_bounds__(256) void gate_softtop(const float* __restrict__ logits,
                                                    float* __restrict__ out_idx,
                                                    float* __restrict__ out_w,
                                                    float* __restrict__ score_sum,
                                                    float* __restrict__ ce_cnt) {
  __shared__ float ls[NEXP];
  __shared__ float lc[NEXP];
  if (threadIdx.x < NEXP) { ls[threadIdx.x] = 0.f; lc[threadIdx.x] = 0.f; }
  __syncthreads();

  const int lane = threadIdx.x & 31;
  const int wave = threadIdx.x >> 5;
  const int t    = blockIdx.x * 8 + wave;

  float s0 = logits[(size_t)t * NEXP + lane];
  float s1 = logits[(size_t)t * NEXP + 32 + lane];

  float m = fmaxf(s0, s1);
#pragma unroll
  for (int sh = 16; sh > 0; sh >>= 1) m = fmaxf(m, __shfl_xor(m, sh, 32));
  float p0 = __expf(s0 - m), p1 = __expf(s1 - m);
  float sum = p0 + p1;
#pragma unroll
  for (int sh = 16; sh > 0; sh >>= 1) sum += __shfl_xor(sum, sh, 32);
  const float inv = 1.f / sum;
  const float sc0 = p0 * inv, sc1 = p1 * inv;

  atomicAdd(&ls[lane], sc0);
  atomicAdd(&ls[lane + 32], sc1);

  float v0 = sc0, v1 = sc1;
  float tw[TOPK]; int ti[TOPK]; float tsum = 0.f;
#pragma unroll
  for (int k = 0; k < TOPK; ++k) {
    float bv = (v1 > v0) ? v1 : v0;          // prefer lower index on ties
    int   bi = (v1 > v0) ? (lane + 32) : lane;
#pragma unroll
    for (int sh = 16; sh > 0; sh >>= 1) {
      float ov = __shfl_xor(bv, sh, 32);
      int   oi = __shfl_xor(bi, sh, 32);
      if (ov > bv || (ov == bv && oi < bi)) { bv = ov; bi = oi; }
    }
    tw[k] = bv; ti[k] = bi; tsum += bv;
    if (bi == lane)      v0 = -1.f;
    if (bi == lane + 32) v1 = -1.f;
  }

  if (lane == 0) {
    const float dn = 1.f / (tsum + 1e-20f);
#pragma unroll
    for (int k = 0; k < TOPK; ++k) {
      out_idx[(size_t)t * TOPK + k] = (float)ti[k];
      out_w  [(size_t)t * TOPK + k] = tw[k] * dn;
      atomicAdd(&lc[ti[k]], 1.f);
    }
  }
  __syncthreads();

  if (threadIdx.x < NEXP) {
    const int b = (blockIdx.x * 8) / SEQLEN;   // 8 tokens never straddle a batch
    atomicAdd(&score_sum[b * NEXP + threadIdx.x], ls[threadIdx.x]);
    atomicAdd(&ce_cnt   [b * NEXP + threadIdx.x], lc[threadIdx.x]);
  }
}

__global__ __launch_bounds__(256) void gate_aux(const float* __restrict__ score_sum,
                                                const float* __restrict__ ce_cnt,
                                                float* __restrict__ out_aux) {
  __shared__ float red[NBATCH * NEXP];
  const int i = threadIdx.x;                      // i -> (b = i/64, e = i%64)
  const float ce = ce_cnt[i] * ((float)NEXP / ((float)SEQLEN * (float)TOPK));
  const float ms = score_sum[i] / (float)SEQLEN;
  red[i] = ce * ms;
  __syncthreads();
  for (int s = (NBATCH * NEXP) / 2; s > 0; s >>= 1) {
    if (i < s) red[i] += red[i + s];
    __syncthreads();
  }
  if (i == 0) out_aux[0] = red[0] * (1.f / (float)NBATCH) * ALPHA;
}

extern "C" void kernel_launch(void* const* d_in, const int* in_sizes, int n_in,
                              void* d_out, int out_size, void* d_ws, size_t ws_size,
                              hipStream_t stream) {
  (void)in_sizes; (void)n_in; (void)out_size; (void)ws_size;
  const float* x = (const float*)d_in[0];   // [4,4096,2048] f32
  const float* w = (const float*)d_in[1];   // [64,2048]     f32
  float* out = (float*)d_out;               // idx[T*6] ++ weight[T*6] ++ aux[1]

  // Workspace layout (all offsets 32B-aligned):
  float*  logits = (float*)d_ws;                                  // 4 MB
  __bf16* wp_hi  = (__bf16*)((char*)d_ws + (size_t)T_TOK * NEXP * 4); // 256 KB
  __bf16* wp_lo  = wp_hi + (size_t)NKCHUNK * 4 * BFRAG;               // 256 KB
  float*  score_sm = (float*)(wp_lo + (size_t)NKCHUNK * 4 * BFRAG);   // 256 f32
  float*  ce_cnt   = score_sm + NBATCH * NEXP;                        // 256 f32

  gate_init<<<2, 256, 0, stream>>>(score_sm);                 // zero both stat arrays
  gate_pack_w<<<(NKCHUNK * 4 * 32) / 256, 256, 0, stream>>>(w, wp_hi, wp_lo);
  gate_gemm_wmma<<<T_TOK / 32 / 4, 128, 0, stream>>>(x, wp_hi, wp_lo, logits);
  gate_softtop<<<T_TOK / 8, 256, 0, stream>>>(logits,
                                              out,
                                              out + (size_t)T_TOK * TOPK,
                                              score_sm, ce_cnt);
  gate_aux<<<1, NBATCH * NEXP, 0, stream>>>(score_sm, ce_cnt,
                                            out + (size_t)2 * T_TOK * TOPK);
}
